// BayesGATLayer_13872744366439
// MI455X (gfx1250) — compile-verified, matrix operand order checked by reading, not run on hardware
//
#include <hip/hip_runtime.h>

typedef __attribute__((ext_vector_type(16))) _Float16 v16h;
typedef __attribute__((ext_vector_type(8)))  _Float16 v8h;
typedef __attribute__((ext_vector_type(4)))  _Float16 v4h;
typedef __attribute__((ext_vector_type(8)))  float    v8f;
typedef __attribute__((ext_vector_type(4)))  float    v4f;

#define N_NODES 50000
#define K_NBR   16
#define DIM     128
#define TAU_F   0.08838834764831845f   // 1/sqrt(128)

// ---------------------------------------------------------------------------
// Kernel 1: convert Wq/Wk/Wv (f32, [k][n] row-major) -> f16 transposed [n][k].
// Makes WMMA B-fragment loads contiguous per lane.
// ---------------------------------------------------------------------------
__global__ void prep_weights_kernel(const float* __restrict__ Wq,
                                    const float* __restrict__ Wk,
                                    const float* __restrict__ Wv,
                                    _Float16* __restrict__ WT) {
    const int k   = threadIdx.x;   // 0..127
    const int n   = blockIdx.x;    // 0..127
    const int mat = blockIdx.y;    // 0..2
    const float* W = (mat == 0) ? Wq : (mat == 1) ? Wk : Wv;
    WT[((size_t)mat * DIM + n) * DIM + k] = (_Float16)W[k * DIM + n];
}

// ---------------------------------------------------------------------------
// WMMA helper: one 16x16 f32 tile = sum of four 16x16x32 f16 WMMAs (K=128).
// A fragments precomputed by caller; B read from f16 W^T (row = output col n).
// Fragment K-order per ISA 7.12.2 (16-bit 16x32): lane-half h covers
// K = kb..kb+7 and kb+16..kb+23 with kb = 32s + 8h.
// ---------------------------------------------------------------------------
__device__ __forceinline__ v8f wmma_tile_f16(const v16h a[4],
                                             const _Float16* __restrict__ wrow,
                                             int half) {
    v8f acc = {};
#pragma unroll
    for (int s = 0; s < 4; ++s) {
        const int kb = s * 32 + half * 8;
        v8h b0 = *(const v8h*)(wrow + kb);
        v8h b1 = *(const v8h*)(wrow + kb + 16);
        v16h bf;
#pragma unroll
        for (int j = 0; j < 8; ++j) { bf[j] = b0[j]; bf[8 + j] = b1[j]; }
        // emits v_wmma_f32_16x16x32_f16
        acc = __builtin_amdgcn_wmma_f32_16x16x32_f16(
            /*neg_a=*/false, a[s], /*neg_b=*/false, bf,
            /*c_mod=*/(short)0, acc, /*reuse_a=*/false, /*reuse_b=*/false);
    }
    return acc;
}

// ---------------------------------------------------------------------------
// Kernel 2: q = z@Wq+bq (f32), k = z@Wk+bk (f16), v = z@Wv+bv (f16).
// 256 threads = 8 waves; block = 16-row tile of z (50000 = 3125*16 exact,
// so EXEC is all-ones, as WMMA requires). Wave w owns column tile [16w,16w+16).
// A fragments (z rows, f32->f16) are built once and reused for all 3 GEMMs.
// ---------------------------------------------------------------------------
__global__ __launch_bounds__(256)
void proj_kernel(const float* __restrict__ z,
                 const _Float16* __restrict__ WT,   // [3][128][128], n-major
                 const float* __restrict__ bq,
                 const float* __restrict__ bk,
                 const float* __restrict__ bv,
                 float*    __restrict__ qbuf,       // [N,128] f32
                 _Float16* __restrict__ khb,        // [N,128] f16
                 _Float16* __restrict__ vhb) {      // [N,128] f16
    const int tile = blockIdx.x;            // 0..3124
    const int wave = threadIdx.x >> 5;      // 0..7  -> column tile
    const int lane = threadIdx.x & 31;
    const int m    = lane & 15;
    const int half = lane >> 4;
    const int row  = tile * 16 + m;         // A row owned by this lane
    const int ncol = wave * 16 + m;         // C/D column owned by this lane
    const int rbase = tile * 16 + 8 * half; // C/D rows: rbase + v, v=0..7

    // Build A fragments (16x32 f16 layout, ISA 7.12.2).
    v16h a[4];
    const float* zr = z + (size_t)row * DIM;
#pragma unroll
    for (int s = 0; s < 4; ++s) {
        const int kb = s * 32 + half * 8;
        v4f f0 = *(const v4f*)(zr + kb);
        v4f f1 = *(const v4f*)(zr + kb + 4);
        v4f f2 = *(const v4f*)(zr + kb + 16);
        v4f f3 = *(const v4f*)(zr + kb + 20);
#pragma unroll
        for (int j = 0; j < 4; ++j) {
            a[s][j]      = (_Float16)f0[j];
            a[s][4 + j]  = (_Float16)f1[j];
            a[s][8 + j]  = (_Float16)f2[j];
            a[s][12 + j] = (_Float16)f3[j];
        }
    }

    // Q projection (store f32)
    {
        v8f acc = wmma_tile_f16(a, WT + (size_t)(0 * DIM + ncol) * DIM, half);
        const float bias = bq[ncol];
#pragma unroll
        for (int v = 0; v < 8; ++v)
            qbuf[(size_t)(rbase + v) * DIM + ncol] = acc[v] + bias;
    }
    // K projection (store f16 to halve gather traffic)
    {
        v8f acc = wmma_tile_f16(a, WT + (size_t)(1 * DIM + ncol) * DIM, half);
        const float bias = bk[ncol];
#pragma unroll
        for (int v = 0; v < 8; ++v)
            khb[(size_t)(rbase + v) * DIM + ncol] = (_Float16)(acc[v] + bias);
    }
    // V projection (store f16)
    {
        v8f acc = wmma_tile_f16(a, WT + (size_t)(2 * DIM + ncol) * DIM, half);
        const float bias = bv[ncol];
#pragma unroll
        for (int v = 0; v < 8; ++v)
            vhb[(size_t)(rbase + v) * DIM + ncol] = (_Float16)(acc[v] + bias);
    }
}

// ---------------------------------------------------------------------------
// Kernel 3: mailbox attention. One wave32 per dst node (50000 = 6250*8).
// Lane covers 4 of 128 dims; per-neighbor dot via butterfly shfl_xor;
// replicated in-wave softmax over 16; f32 accumulation of f16 values.
// ---------------------------------------------------------------------------
__global__ __launch_bounds__(256)
void attn_kernel(const float* __restrict__ qbuf,
                 const _Float16* __restrict__ khb,
                 const _Float16* __restrict__ vhb,
                 const int* __restrict__ src,
                 float* __restrict__ out_h,       // [N,128]
                 float* __restrict__ out_alpha) { // [N,16]
    const int wave = threadIdx.x >> 5;
    const int lane = threadIdx.x & 31;
    const int node = blockIdx.x * 8 + wave;

    const v4f qv = *(const v4f*)(qbuf + (size_t)node * DIM + lane * 4);

    const int4* s4 = (const int4*)(src + (size_t)node * K_NBR);
    int4 si0 = s4[0], si1 = s4[1], si2 = s4[2], si3 = s4[3];
    const int sidx[16] = { si0.x, si0.y, si0.z, si0.w,
                           si1.x, si1.y, si1.z, si1.w,
                           si2.x, si2.y, si2.z, si2.w,
                           si3.x, si3.y, si3.z, si3.w };

    float e[16];
#pragma unroll
    for (int k = 0; k < 16; ++k) {
        v4h kv = *(const v4h*)(khb + (size_t)sidx[k] * DIM + lane * 4);
        float d = qv[0] * (float)kv[0] + qv[1] * (float)kv[1]
                + qv[2] * (float)kv[2] + qv[3] * (float)kv[3];
#pragma unroll
        for (int off = 16; off > 0; off >>= 1)
            d += __shfl_xor(d, off, 32);       // all lanes end with full dot
        e[k] = d * TAU_F;
    }

    // softmax over 16 (values replicated across the wave)
    float mx = e[0];
#pragma unroll
    for (int k = 1; k < 16; ++k) mx = fmaxf(mx, e[k]);
    float sum = 0.0f;
#pragma unroll
    for (int k = 0; k < 16; ++k) { e[k] = __expf(e[k] - mx); sum += e[k]; }
    const float inv = 1.0f / sum;

    v4f acc = {0.0f, 0.0f, 0.0f, 0.0f};
#pragma unroll
    for (int k = 0; k < 16; ++k) {
        const float al = e[k] * inv;
        v4h vv = *(const v4h*)(vhb + (size_t)sidx[k] * DIM + lane * 4);
        acc[0] += al * (float)vv[0];
        acc[1] += al * (float)vv[1];
        acc[2] += al * (float)vv[2];
        acc[3] += al * (float)vv[3];
    }
    *(v4f*)(out_h + (size_t)node * DIM + lane * 4) = acc;

    // lane l (<16) writes alpha[l]; select-chain keeps e[] in registers
    if (lane < 16) {
        float myAl = 0.0f;
#pragma unroll
        for (int k = 0; k < 16; ++k)
            myAl = (lane == k) ? e[k] * inv : myAl;
        out_alpha[(size_t)node * K_NBR + lane] = myAl;
    }
}

// ---------------------------------------------------------------------------
// Launch: inputs per setup_inputs() order:
//   0:z 1:src 2:Wq 3:bq 4:Wk 5:bk 6:Wv 7:bv 8:Ws1 9:bs1 10:Ws2 11:bs2
// (SE path Ws1/Ws2 has no effect on the returned (h, alpha) -> skipped.)
// d_out = h [N*128] f32 ++ alpha [N*16] f32.
// ---------------------------------------------------------------------------
extern "C" void kernel_launch(void* const* d_in, const int* in_sizes, int n_in,
                              void* d_out, int out_size, void* d_ws, size_t ws_size,
                              hipStream_t stream) {
    const float* z  = (const float*)d_in[0];
    const int* src  = (const int*)d_in[1];
    const float* Wq = (const float*)d_in[2];
    const float* bq = (const float*)d_in[3];
    const float* Wk = (const float*)d_in[4];
    const float* bk = (const float*)d_in[5];
    const float* Wv = (const float*)d_in[6];
    const float* bv = (const float*)d_in[7];

    float* out_h     = (float*)d_out;
    float* out_alpha = out_h + (size_t)N_NODES * DIM;

    // Workspace layout (~51.4 MB, all 16B aligned):
    char* ws = (char*)d_ws;
    _Float16* WT  = (_Float16*)ws;                               // 98304 B
    float*    qb  = (float*)(ws + 98304);                        // 25.6 MB
    _Float16* khb = (_Float16*)(ws + 98304 + (size_t)N_NODES * DIM * 4);
    _Float16* vhb = khb + (size_t)N_NODES * DIM;                 // +12.8 MB each

    prep_weights_kernel<<<dim3(DIM, 3), DIM, 0, stream>>>(Wq, Wk, Wv, WT);
    proj_kernel<<<N_NODES / 16, 256, 0, stream>>>(z, WT, bq, bk, bv, qb, khb, vhb);
    attn_kernel<<<N_NODES / 8, 256, 0, stream>>>(qb, khb, vhb, src, out_h, out_alpha);
}
